// W8A8Linear_90761248899143
// MI455X (gfx1250) — matrix operand order repaired and checked by reading, not compiled, not run on hardware
//
#include <hip/hip_runtime.h>
#include <hip/hip_fp16.h>

typedef __attribute__((ext_vector_type(16))) _Float16 v16h;
typedef __attribute__((ext_vector_type(32))) _Float16 v32h;
typedef __attribute__((ext_vector_type(8)))  _Float16 v8h;
typedef __attribute__((ext_vector_type(4)))  _Float16 v4h;
typedef __attribute__((ext_vector_type(8)))  float    v8f;

#define EPS 1e-5f
#define QMAX 127.0f

#ifndef __has_builtin
#define __has_builtin(x) 0
#endif
#if __has_builtin(__builtin_amdgcn_swmmac_f32_16x16x64_f16)
#define USE_SWMMAC 1
#else
#define USE_SWMMAC 0
#endif

union AV { v16h v; v8h h[2]; };
union BV { v32h v; v8h h[4]; };

// ---------------- Kernel 1: column L2 norm of W ----------------
__global__ __launch_bounds__(256) void colnorm_kernel(const float* __restrict__ w,
                                                      float* __restrict__ colnorm,
                                                      int O, int F) {
    int f = blockIdx.x * blockDim.x + threadIdx.x;
    if (f >= F) return;
    float acc = 0.f;
    for (int r = 0; r < O; ++r) {
        float v = w[(size_t)r * F + f];
        acc = fmaf(v, v, acc);
    }
    colnorm[f] = sqrtf(acc);
}

// ---------------- Kernel 2: global min + per-feature scale ----------------
__global__ __launch_bounds__(1024) void scale_kernel(const float* __restrict__ colnorm,
                                                     float* __restrict__ scale, int F) {
    __shared__ float red[1024];
    float m = 3.4e38f;
    for (int f = threadIdx.x; f < F; f += blockDim.x) m = fminf(m, colnorm[f]);
    red[threadIdx.x] = m;
    __syncthreads();
    for (int s = 512; s > 0; s >>= 1) {
        if (threadIdx.x < (unsigned)s)
            red[threadIdx.x] = fminf(red[threadIdx.x], red[threadIdx.x + s]);
        __syncthreads();
    }
    float denom = fmaxf(red[0], EPS);
    for (int f = threadIdx.x; f < F; f += blockDim.x) scale[f] = colnorm[f] / denom;
}

// ---------------- Kernel 3: W fp32 -> fp16 ----------------
__global__ __launch_bounds__(256) void wcvt_kernel(const float* __restrict__ w,
                                                   _Float16* __restrict__ wh, size_t n) {
    size_t i = ((size_t)blockIdx.x * blockDim.x + threadIdx.x) * 4;
    if (i >= n) return;
    float4 v = *(const float4*)(w + i);
    v4h h;
    h.x = (_Float16)v.x; h.y = (_Float16)v.y; h.z = (_Float16)v.z; h.w = (_Float16)v.w;
    *(v4h*)(wh + i) = h;
}

// ---------------- Kernel 4: 2:4 sparsify + per-token int-grid quant + pack ----------
// One block per token: 256 threads x 16 features = 4096 features.
// SWMMAC path: writes packed A (2 kept values/group, ascending index) and 4-bit/group
// index nibbles packed 8-groups-per-u32 (== one index VGPR per lane per K=64 block).
__global__ __launch_bounds__(256) void sq_kernel(const float* __restrict__ x,
                                                 const float* __restrict__ scale,
                                                 _Float16* __restrict__ qout,   // packed or dense
                                                 unsigned* __restrict__ idxb,   // [T, F/32] (swmmac only)
                                                 float* __restrict__ stok, int F) {
    __shared__ float red[256];
#if USE_SWMMAC
    __shared__ unsigned short nib[256];
#endif
    const int t = blockIdx.x;
    const float* xt = x + (size_t)t * F;
    const int f0 = threadIdx.x * 16;

    float v[16];
    int k0g[4], k1g[4];
    float amax = 0.f;
#pragma unroll
    for (int g = 0; g < 4; ++g) {
        const int fb = f0 + g * 4;
        float4 xv = *(const float4*)(xt + fb);
        float4 sv = *(const float4*)(scale + fb);
        v[g*4+0] = xv.x; v[g*4+1] = xv.y; v[g*4+2] = xv.z; v[g*4+3] = xv.w;
        float m[4] = { fabsf(xv.x)*sv.x, fabsf(xv.y)*sv.y,
                       fabsf(xv.z)*sv.z, fabsf(xv.w)*sv.w };
        // stable argmin (first occurrence) == argsort()[0]
        int i0 = 0; float b0 = m[0];
#pragma unroll
        for (int j = 1; j < 4; ++j) { if (m[j] < b0) { b0 = m[j]; i0 = j; } }
        // stable second-smallest among the rest == argsort()[1]
        int i1 = 0; float b1 = 3.4e38f;
#pragma unroll
        for (int j = 0; j < 4; ++j) {
            bool take = (j != i0) && (m[j] < b1);
            b1 = take ? m[j] : b1;
            i1 = take ? j    : i1;
        }
        // kept (survivor) indices in ascending order -> Idx0 < Idx1 rule
        int k0 = -1, k1 = -1;
#pragma unroll
        for (int j = 0; j < 4; ++j) {
            bool keep = (j != i0) && (j != i1);
            k1 = (keep && k0 >= 0) ? j : k1;
            k0 = (keep && k0 < 0)  ? j : k0;
        }
        k0g[g] = k0; k1g[g] = k1;
#pragma unroll
        for (int j = 0; j < 4; ++j) {
            bool kill = (j == i0) || (j == i1);
            v[g*4+j] = kill ? 0.f : v[g*4+j];
            amax = fmaxf(amax, fabsf(v[g*4+j]));
        }
    }
    red[threadIdx.x] = amax;
    __syncthreads();
    for (int s = 128; s > 0; s >>= 1) {
        if (threadIdx.x < (unsigned)s)
            red[threadIdx.x] = fmaxf(red[threadIdx.x], red[threadIdx.x + s]);
        __syncthreads();
    }
    const float sq = fmaxf(red[0], EPS) / QMAX;
    if (threadIdx.x == 0) stok[t] = sq;
    const float inv = 1.f / sq;

#if USE_SWMMAC
    // Packed values: compressed positions 2g, 2g+1 hold survivors (ascending index)
    v8h pk;
    unsigned short nb = 0;
#pragma unroll
    for (int g = 0; g < 4; ++g) {
        const int k0 = k0g[g], k1 = k1g[g];
        float g0 = v[g*4+0], g1 = v[g*4+1], g2 = v[g*4+2], g3 = v[g*4+3];
        float vk0 = (k0 == 0) ? g0 : (k0 == 1) ? g1 : (k0 == 2) ? g2 : g3;
        float vk1 = (k1 == 1) ? g1 : (k1 == 2) ? g2 : g3;
        pk[g*2+0] = (_Float16)rintf(vk0 * inv);   // integer in [-127,127]: exact fp16
        pk[g*2+1] = (_Float16)rintf(vk1 * inv);
        nb |= (unsigned short)((k0 | (k1 << 2)) << (g * 4));
    }
    *(v8h*)(qout + (size_t)t * (F/2) + threadIdx.x * 8) = pk;
    nib[threadIdx.x] = nb;
    __syncthreads();
    if (threadIdx.x < (unsigned)(F/32)) {  // F/32 u32 words per token
        unsigned word = (unsigned)nib[threadIdx.x*2] | ((unsigned)nib[threadIdx.x*2+1] << 16);
        idxb[(size_t)t * (F/32) + threadIdx.x] = word;
    }
#else
    _Float16* qt = qout + (size_t)t * F + f0;
#pragma unroll
    for (int g = 0; g < 4; ++g) {
        v4h h;
        h.x = (_Float16)rintf(v[g*4+0] * inv);
        h.y = (_Float16)rintf(v[g*4+1] * inv);
        h.z = (_Float16)rintf(v[g*4+2] * inv);
        h.w = (_Float16)rintf(v[g*4+3] * inv);
        *(v4h*)(qt + g * 4) = h;
    }
#endif
}

#if USE_SWMMAC
// ---------------- Kernel 5 (SWMMAC): y = s[t] * (sparse_q @ Wh^T) + bias ----------------
// 8 waves (4Mx2N); wave tile 32x32; K-step 64 -> 4 v_swmmac per iteration.
__global__ __launch_bounds__(256) void gemm_swmmac_kernel(const _Float16* __restrict__ qp,
                                                          const unsigned* __restrict__ idxb,
                                                          const _Float16* __restrict__ wh,
                                                          const float* __restrict__ stok,
                                                          const float* __restrict__ bias,
                                                          float* __restrict__ out,
                                                          int T, int O, int F) {
    const int lane  = threadIdx.x & 31;
    const int wave  = threadIdx.x >> 5;
    const int wm    = wave & 3;
    const int wn    = wave >> 2;
    const int mBase = blockIdx.x * 128 + wm * 32;
    const int nBase = blockIdx.y * 64  + wn * 32;
    const int lhalf = lane >> 4;
    const int lmod  = lane & 15;
    const int Fc = F / 2;    // compressed K per token
    const int NW = F / 32;   // index words per token

    // Packed A (stored 16x32 layout): lane row m=lmod; halves 0-7 = Kc[off..off+7],
    // halves 8-15 = Kc[off+16..off+23], off = lhalf*8.
    const _Float16* arow0 = qp + (size_t)(mBase + lmod)      * Fc + lhalf * 8;
    const _Float16* arow1 = qp + (size_t)(mBase + 16 + lmod) * Fc + lhalf * 8;
    // Index VGPR: lane row m=lmod, groups (lhalf*8..+7) of K=64 block kb -> word kb*2+lhalf
    const unsigned* irow0 = idxb + (size_t)(mBase + lmod)      * NW + lhalf;
    const unsigned* irow1 = idxb + (size_t)(mBase + 16 + lmod) * NW + lhalf;
    // B (64x16): lane col n=lmod; halves 0-15 = K[lhalf*16..+15], 16-31 = K[32+lhalf*16..+15]
    const _Float16* brow0 = wh + (size_t)(nBase + lmod)      * F + lhalf * 16;
    const _Float16* brow1 = wh + (size_t)(nBase + 16 + lmod) * F + lhalf * 16;

    v8f acc00 = {}, acc01 = {}, acc10 = {}, acc11 = {};

    for (int kb = 0; kb < F / 64; ++kb) {
        const int ck = kb * 32;
        const int k  = kb * 64;
        AV a0, a1; BV b0, b1;
        a0.h[0] = *(const v8h*)(arow0 + ck);
        a0.h[1] = *(const v8h*)(arow0 + ck + 16);
        a1.h[0] = *(const v8h*)(arow1 + ck);
        a1.h[1] = *(const v8h*)(arow1 + ck + 16);
        const int idxA0 = (int)irow0[kb * 2];
        const int idxA1 = (int)irow1[kb * 2];
        b0.h[0] = *(const v8h*)(brow0 + k);
        b0.h[1] = *(const v8h*)(brow0 + k + 8);
        b0.h[2] = *(const v8h*)(brow0 + k + 32);
        b0.h[3] = *(const v8h*)(brow0 + k + 40);
        b1.h[0] = *(const v8h*)(brow1 + k);
        b1.h[1] = *(const v8h*)(brow1 + k + 8);
        b1.h[2] = *(const v8h*)(brow1 + k + 32);
        b1.h[3] = *(const v8h*)(brow1 + k + 40);

        acc00 = __builtin_amdgcn_swmmac_f32_16x16x64_f16(false, a0.v, false, b0.v,
                                                         acc00, idxA0, false, false);
        acc01 = __builtin_amdgcn_swmmac_f32_16x16x64_f16(false, a0.v, false, b1.v,
                                                         acc01, idxA0, false, false);
        acc10 = __builtin_amdgcn_swmmac_f32_16x16x64_f16(false, a1.v, false, b0.v,
                                                         acc10, idxA1, false, false);
        acc11 = __builtin_amdgcn_swmmac_f32_16x16x64_f16(false, a1.v, false, b1.v,
                                                         acc11, idxA1, false, false);
    }

    const v8f accs[2][2] = { { acc00, acc01 }, { acc10, acc11 } };
#pragma unroll
    for (int j = 0; j < 2; ++j) {
        const int n = nBase + j * 16 + lmod;
        const float bv = bias[n];
#pragma unroll
        for (int i = 0; i < 2; ++i) {
            const int m0 = mBase + i * 16 + lhalf * 8;
#pragma unroll
            for (int r = 0; r < 8; ++r) {
                const int m = m0 + r;
                out[(size_t)m * O + n] = fmaf(stok[m], accs[i][j][r], bv);
            }
        }
    }
}
#else
// ---------------- Kernel 5 (dense WMMA fallback) ----------------
__global__ __launch_bounds__(256) void gemm_kernel(const _Float16* __restrict__ qx,
                                                   const _Float16* __restrict__ wh,
                                                   const float* __restrict__ stok,
                                                   const float* __restrict__ bias,
                                                   float* __restrict__ out,
                                                   int T, int O, int F) {
    const int lane  = threadIdx.x & 31;
    const int wave  = threadIdx.x >> 5;
    const int wm    = wave & 3;
    const int wn    = wave >> 2;
    const int mBase = blockIdx.x * 128 + wm * 32;
    const int nBase = blockIdx.y * 64  + wn * 32;
    const int lhalf = lane >> 4;
    const int lmod  = lane & 15;

    const _Float16* arow0 = qx + (size_t)(mBase + lmod)      * F + lhalf * 8;
    const _Float16* arow1 = qx + (size_t)(mBase + 16 + lmod) * F + lhalf * 8;
    const _Float16* brow0 = wh + (size_t)(nBase + lmod)      * F + lhalf * 8;
    const _Float16* brow1 = wh + (size_t)(nBase + 16 + lmod) * F + lhalf * 8;

    v8f acc00 = {}, acc01 = {}, acc10 = {}, acc11 = {};

    for (int k = 0; k < F; k += 32) {
        AV a0, a1, b0, b1;
        a0.h[0] = *(const v8h*)(arow0 + k);
        a0.h[1] = *(const v8h*)(arow0 + k + 16);
        a1.h[0] = *(const v8h*)(arow1 + k);
        a1.h[1] = *(const v8h*)(arow1 + k + 16);
        b0.h[0] = *(const v8h*)(brow0 + k);
        b0.h[1] = *(const v8h*)(brow0 + k + 16);
        b1.h[0] = *(const v8h*)(brow1 + k);
        b1.h[1] = *(const v8h*)(brow1 + k + 16);

        acc00 = __builtin_amdgcn_wmma_f32_16x16x32_f16(false, a0.v, false, b0.v,
                                                       (short)0, acc00, false, false);
        acc01 = __builtin_amdgcn_wmma_f32_16x16x32_f16(false, a0.v, false, b1.v,
                                                       (short)0, acc01, false, false);
        acc10 = __builtin_amdgcn_wmma_f32_16x16x32_f16(false, a1.v, false, b0.v,
                                                       (short)0, acc10, false, false);
        acc11 = __builtin_amdgcn_wmma_f32_16x16x32_f16(false, a1.v, false, b1.v,
                                                       (short)0, acc11, false, false);
    }

    const v8f accs[2][2] = { { acc00, acc01 }, { acc10, acc11 } };
#pragma unroll
    for (int j = 0; j < 2; ++j) {
        const int n = nBase + j * 16 + lmod;
        const float bv = bias[n];
#pragma unroll
        for (int i = 0; i < 2; ++i) {
            const int m0 = mBase + i * 16 + lhalf * 8;
#pragma unroll
            for (int r = 0; r < 8; ++r) {
                const int m = m0 + r;
                out[(size_t)m * O + n] = fmaf(stok[m], accs[i][j][r], bv);
            }
        }
    }
}
#endif

// ---------------- Launch ----------------
extern "C" void kernel_launch(void* const* d_in, const int* in_sizes, int n_in,
                              void* d_out, int out_size, void* d_ws, size_t ws_size,
                              hipStream_t stream) {
    const float* x    = (const float*)d_in[0];
    const float* w    = (const float*)d_in[1];
    const float* bias = (const float*)d_in[2];
    float* out = (float*)d_out;

    const int F = 4096;
    const int O = in_sizes[1] / F;
    const int T = in_sizes[0] / F;

    // Workspace layout: [region A: dense qx (T*F*2B) | packed qp (T*F B) + idx (T*F/8 B)]
    //                   [wh: O*F*2B] [stats]
    char* ws = (char*)d_ws;
    _Float16* qregion = (_Float16*)ws;
    unsigned* idxb    = (unsigned*)(ws + (size_t)T * F);          // after packed qp
    _Float16* wh      = (_Float16*)(ws + (size_t)T * F * 2);
    float* colnorm    = (float*)(ws + (size_t)T * F * 2 + (size_t)O * F * 2);
    float* scalev     = colnorm + F;
    float* stok       = scalev + F;

    colnorm_kernel<<<(F + 255) / 256, 256, 0, stream>>>(w, colnorm, O, F);
    scale_kernel<<<1, 1024, 0, stream>>>(colnorm, scalev, F);
    {
        size_t n = (size_t)O * F;
        wcvt_kernel<<<(unsigned)((n / 4 + 255) / 256), 256, 0, stream>>>(w, wh, n);
    }
    sq_kernel<<<T, 256, 0, stream>>>(x, scalev, qregion, idxb, stok, F);
#if USE_SWMMAC
    gemm_swmmac_kernel<<<dim3(T / 128, O / 64), 256, 0, stream>>>(qregion, idxb, wh,
                                                                  stok, bias, out, T, O, F);
#else
    gemm_kernel<<<dim3(T / 128, O / 64), 256, 0, stream>>>(qregion, wh, stok, bias, out, T, O, F);
#endif
}
